// ManualCausalSelfAttention_88407606821229
// MI455X (gfx1250) — compile-verified, matrix operand order checked by reading, not run on hardware
//
#include <hip/hip_runtime.h>
#include <hip/hip_bf16.h>

// ---------------- problem constants ----------------
#define B_  4
#define S_  2048
#define D_  1024
#define H_  16
#define HD_ 64

typedef __attribute__((ext_vector_type(16))) __bf16        v16bf;
typedef __attribute__((ext_vector_type(8)))  float         v8f;
typedef __attribute__((ext_vector_type(4)))  unsigned int  uint4v;

// float -> bf16, round-to-nearest-even
__device__ __forceinline__ unsigned short f2bf(float f) {
  unsigned int u = __float_as_uint(f);
  u += 0x7FFFu + ((u >> 16) & 1u);
  return (unsigned short)(u >> 16);
}

// WMMA fragment: 16 bf16 values per lane (8 VGPRs)
union Frag {
  v16bf          bf;
  uint4v         q[2];
  unsigned int   u32[8];
  unsigned short u16[16];
};
union Acc { v8f v; float f[8]; };

// Load a 16-bit A or B fragment. Per ISA (CDNA5 ISA, 16-bit A 16x32):
// lanes 0-15 hold K {0..7,16..23}, lanes 16-31 hold K {8..15,24..31};
// each lane's 16 bf16 = two 16B chunks. Source layout row*ld + k (k contiguous).
__device__ __forceinline__ void load_frag(Frag& f, const unsigned short* base,
                                          int row, int ld, int kbase) {
  const int lane = threadIdx.x & 31;
  const unsigned short* p = base + (size_t)row * ld + kbase + ((lane & 16) >> 1);
  f.q[0] = *(const uint4v*)(p);
  f.q[1] = *(const uint4v*)(p + 16);
}

__device__ __forceinline__ v8f wmma_bf16(const Frag& a, const Frag& b, v8f c) {
  return __builtin_amdgcn_wmma_f32_16x16x32_bf16(
      false, a.bf, false, b.bf, (short)0, c, false, false);
}

// CDNA5 async copy: global -> LDS, 16B per lane, tracked by ASYNCcnt.
// lds_off = low 32 bits of generic shared pointer (flat LDS addr truncates
// to addr[31:0] per ISA 10.2), gaddr = 64-bit global address.
__device__ __forceinline__ void async_copy_b128(void* lds_ptr, const void* gptr) {
  unsigned lds_off = (unsigned)(size_t)lds_ptr;
  unsigned long long ga = (unsigned long long)(size_t)gptr;
  asm volatile("global_load_async_to_lds_b128 %0, %1, off"
               :: "v"(lds_off), "v"(ga) : "memory");
}
__device__ __forceinline__ void wait_async0() {
  asm volatile("s_wait_asynccnt 0" ::: "memory");
}

// ---------------- fp32 -> bf16 conversion ----------------
__global__ void cvt_bf16(const float* __restrict__ src,
                         unsigned short* __restrict__ dst, int n) {
  int i = blockIdx.x * blockDim.x + threadIdx.x;
  if (i < n) dst[i] = f2bf(src[i]);
}

// ---------------- tiled bf16 GEMM: C[M=8192, Ngl] = A[M,1024] @ W[1024,Ngl] + bias
// Block tile 256x128, 8 waves, wave tile 64x64 (16 WMMA : 16 LDS-b128 loads).
// mode 0: QKV epilogue (scatter to q/k/v [B,H,S,HD] bf16, q scaled by 0.125)
// mode 1: proj epilogue (f32 out * padding)
__global__ __launch_bounds__(256) void gemm_bf16(
    const unsigned short* __restrict__ Ab, const unsigned short* __restrict__ Bw,
    const float* __restrict__ bias, int Ngl, int mode,
    unsigned short* __restrict__ qb, unsigned short* __restrict__ kb,
    unsigned short* __restrict__ vb, float* __restrict__ out,
    const unsigned char* __restrict__ pad) {
  __shared__ __align__(16) unsigned short As[256 * 32];  // m-major, ld=32
  __shared__ __align__(16) unsigned short Bs[128 * 32];  // n-major, ld=32

  const int tid  = threadIdx.x;
  const int lane = tid & 31;
  const int w    = tid >> 5;
  const int wm   = w & 3;   // wave m-tile (0..3) -> 64 rows each
  const int wn   = w >> 2;  // wave n-tile (0..1) -> 64 cols each
  const int m0   = blockIdx.x * 256;
  const int n0   = blockIdx.y * 128;

  Acc acc[4][4];
#pragma unroll
  for (int mi = 0; mi < 4; mi++)
#pragma unroll
    for (int nj = 0; nj < 4; nj++)
#pragma unroll
      for (int r = 0; r < 8; r++) acc[mi][nj].f[r] = 0.f;

  const int Nh = Ngl >> 1;  // W row length in u32

  for (int k0 = 0; k0 < 1024; k0 += 32) {
    // stage A tile (256x32, row-major): 1024 async b128, 4 per thread
    {
#pragma unroll
      for (int i = 0; i < 4; i++) {
        int j   = tid + i * 256;      // 0..1023 (uint4 index into As)
        int row = j >> 2, c4 = j & 3; // 4 uint4 per 32-elem row
        const uint4v* g = (const uint4v*)(Ab + (size_t)(m0 + row) * 1024 + k0) + c4;
        async_copy_b128(&((uint4v*)As)[j], g);
      }
    }
    // stage B tile transposed to n-major (coalesced u32 reads of W rows)
    {
      const unsigned int* gB = (const unsigned int*)Bw;
#pragma unroll
      for (int i = 0; i < 8; i++) {
        int j  = tid + i * 256;       // 0..2047
        int k  = j >> 6, np = j & 63; // 64 u32 per W-tile row
        unsigned int wv = gB[(size_t)(k0 + k) * Nh + (n0 >> 1) + np];
        int n = np * 2;
        Bs[n * 32 + k]       = (unsigned short)(wv & 0xffffu);
        Bs[(n + 1) * 32 + k] = (unsigned short)(wv >> 16);
      }
    }
    wait_async0();
    __syncthreads();

    Frag af[4], bfr[4];
#pragma unroll
    for (int mi = 0; mi < 4; mi++)
      load_frag(af[mi], As, wm * 64 + mi * 16 + (lane & 15), 32, 0);
#pragma unroll
    for (int nj = 0; nj < 4; nj++)
      load_frag(bfr[nj], Bs, wn * 64 + nj * 16 + (lane & 15), 32, 0);
#pragma unroll
    for (int mi = 0; mi < 4; mi++)
#pragma unroll
      for (int nj = 0; nj < 4; nj++)
        acc[mi][nj].v = wmma_bf16(af[mi], bfr[nj], acc[mi][nj].v);
    __syncthreads();
  }

  // epilogue — C layout: VGPR r holds (m = r, lanes 0-15) and (m = r+8, lanes 16-31)
  const int mhalf = (lane & 16) >> 1;
#pragma unroll
  for (int mi = 0; mi < 4; mi++) {
#pragma unroll
    for (int nj = 0; nj < 4; nj++) {
#pragma unroll
      for (int r = 0; r < 8; r++) {
        int m = m0 + wm * 64 + mi * 16 + r + mhalf;
        int n = n0 + wn * 64 + nj * 16 + (lane & 15);
        float val = acc[mi][nj].f[r] + bias[n];
        if (mode == 0) {
          int sec = n >> 10, dc = n & 1023;
          int h = dc >> 6, hd = dc & 63;
          int b = m >> 11, s = m & 2047;
          size_t off = (((size_t)b * H_ + h) * S_ + s) * HD_ + hd;
          if (sec == 0)       qb[off] = f2bf(val * 0.125f);  // fold 1/sqrt(64)
          else if (sec == 1)  kb[off] = f2bf(val);
          else                vb[off] = f2bf(val);
        } else {
          float pv = pad[m] ? 1.0f : 0.0f;
          out[(size_t)m * 1024 + n] = val * pv;
        }
      }
    }
  }
}

// ---------------- flash attention: block = (64 q rows, one (b,h)); 4 waves x 16 rows
__global__ __launch_bounds__(128) void attn(
    const unsigned short* __restrict__ qb, const unsigned short* __restrict__ kb,
    const unsigned short* __restrict__ vb, const unsigned char* __restrict__ pad,
    unsigned short* __restrict__ yb) {
  __shared__ __align__(16) unsigned short Ks[64 * 64];      // [k][d] row-major
  __shared__ __align__(16) unsigned short Vt[64 * 64];      // [d][k] (transposed)
  __shared__ __align__(16) unsigned short Ps[4][16 * 64];   // per-wave P staging

  const int tid = threadIdx.x, lane = tid & 31, w = tid >> 5;
  const int qt = blockIdx.x, bh = blockIdx.y;
  const int b = bh >> 4, h = bh & 15;
  const size_t bho = (size_t)bh * S_ * HD_;
  const unsigned short* qg = qb + bho;
  const unsigned short* kg = kb + bho;
  const unsigned short* vg = vb + bho;
  const unsigned char*  padr = pad + (size_t)b * S_;
  const int q0 = qt * 64 + w * 16;
  const int mhalf = (lane & 16) >> 1;
  const int ncol  = lane & 15;

  Frag qf[2];
  load_frag(qf[0], qg, q0 + ncol, 64, 0);
  load_frag(qf[1], qg, q0 + ncol, 64, 32);

  float mrow[8], lrow[8];
  Acc accO[4];
#pragma unroll
  for (int r = 0; r < 8; r++) { mrow[r] = -1e30f; lrow[r] = 0.f; }
#pragma unroll
  for (int c = 0; c < 4; c++)
#pragma unroll
    for (int r = 0; r < 8; r++) accO[c].f[r] = 0.f;

  for (int kt = 0; kt <= qt; ++kt) {
    // stage K tile via async copy (contiguous), V tile transposed manually
    {
      const uint4v* gK = (const uint4v*)(kg + (size_t)kt * 64 * 64);
#pragma unroll
      for (int i = 0; i < 4; i++) {
        int j = tid + i * 128;  // 0..511
        async_copy_b128(&((uint4v*)Ks)[j], &gK[j]);
      }
      const unsigned int* gV = (const unsigned int*)(vg + (size_t)kt * 64 * 64);
#pragma unroll
      for (int i = 0; i < 16; i++) {
        int j = tid + i * 128;            // 0..2047
        int k = j >> 5, dp = j & 31;
        unsigned int wv = gV[j];
        int d = dp * 2;
        Vt[d * 64 + k]       = (unsigned short)(wv & 0xffffu);
        Vt[(d + 1) * 64 + k] = (unsigned short)(wv >> 16);
      }
    }
    wait_async0();
    __syncthreads();

    // S = Q K^T  (B fragment: lane n reads K row n, d contiguous)
    Acc sA[4];
#pragma unroll
    for (int j = 0; j < 4; j++) {
      Frag kf;
      v8f c = {0.f, 0.f, 0.f, 0.f, 0.f, 0.f, 0.f, 0.f};
      load_frag(kf, Ks, j * 16 + ncol, 64, 0);
      c = wmma_bf16(qf[0], kf, c);
      load_frag(kf, Ks, j * 16 + ncol, 64, 32);
      c = wmma_bf16(qf[1], kf, c);
      sA[j].v = c;
    }

    // causal + padding mask
#pragma unroll
    for (int j = 0; j < 4; j++) {
      int kgi = kt * 64 + j * 16 + ncol;
      bool pd = padr[kgi] != 0;
#pragma unroll
      for (int r = 0; r < 8; r++) {
        int qgi = q0 + r + mhalf;
        sA[j].f[r] = (pd && kgi <= qgi) ? sA[j].f[r] : -1e30f;
      }
    }

    // online softmax; rows live in one VGPR across a 16-lane half
#pragma unroll
    for (int r = 0; r < 8; r++) {
      float mx = fmaxf(fmaxf(sA[0].f[r], sA[1].f[r]),
                       fmaxf(sA[2].f[r], sA[3].f[r]));
#pragma unroll
      for (int off = 8; off >= 1; off >>= 1)
        mx = fmaxf(mx, __shfl_xor(mx, off, 16));
      float mnew = fmaxf(mrow[r], mx);
      float corr = __expf(mrow[r] - mnew);
      float ps = 0.f;
#pragma unroll
      for (int j = 0; j < 4; j++) {
        float p = __expf(sA[j].f[r] - mnew);
        ps += p;
        Ps[w][(r + mhalf) * 64 + j * 16 + ncol] = f2bf(p);
      }
#pragma unroll
      for (int off = 8; off >= 1; off >>= 1) ps += __shfl_xor(ps, off, 16);
      lrow[r] = lrow[r] * corr + ps;
      mrow[r] = mnew;
#pragma unroll
      for (int c = 0; c < 4; c++) accO[c].f[r] *= corr;
    }

    // wave-local P staging: wave32 lockstep, just drain LDS stores
    asm volatile("s_wait_dscnt 0" ::: "memory");

    // O += P V  (A = P 16x64; B fragment: lane d reads Vt row d, k contiguous)
    Frag pf0, pf1;
    load_frag(pf0, &Ps[w][0], ncol, 64, 0);
    load_frag(pf1, &Ps[w][0], ncol, 64, 32);
#pragma unroll
    for (int c = 0; c < 4; c++) {
      Frag vf;
      load_frag(vf, Vt, c * 16 + ncol, 64, 0);
      accO[c].v = wmma_bf16(pf0, vf, accO[c].v);
      load_frag(vf, Vt, c * 16 + ncol, 64, 32);
      accO[c].v = wmma_bf16(pf1, vf, accO[c].v);
    }
    __syncthreads();
  }

  // normalize + write y (bf16, [B,S,D]) for the proj GEMM
#pragma unroll
  for (int c = 0; c < 4; c++) {
#pragma unroll
    for (int r = 0; r < 8; r++) {
      int m = r + mhalf;
      int s = q0 + m;
      int d = h * 64 + c * 16 + ncol;
      float l = lrow[r];
      float val = (l > 0.f) ? accO[c].f[r] / l : 0.f;
      yb[((size_t)b * S_ + s) * D_ + d] = f2bf(val);
    }
  }
}

// ---------------- launcher ----------------
extern "C" void kernel_launch(void* const* d_in, const int* in_sizes, int n_in,
                              void* d_out, int out_size, void* d_ws, size_t ws_size,
                              hipStream_t stream) {
  const float*         x     = (const float*)d_in[0];
  const unsigned char* pad   = (const unsigned char*)d_in[1];
  const float*         Wqkv  = (const float*)d_in[2];
  const float*         bqkv  = (const float*)d_in[3];
  const float*         Wproj = (const float*)d_in[4];
  const float*         bproj = (const float*)d_in[5];
  float*               out   = (float*)d_out;

  char* ws = (char*)d_ws;
  size_t off = 0;
  auto take = [&](size_t elems) -> unsigned short* {
    unsigned short* p = (unsigned short*)(ws + off);
    off += ((elems * 2 + 255) & ~(size_t)255);
    return p;
  };
  unsigned short* xb     = take((size_t)B_ * S_ * D_);
  unsigned short* wqkvb  = take((size_t)D_ * 3 * D_);
  unsigned short* wprojb = take((size_t)D_ * D_);
  unsigned short* qbuf   = take((size_t)B_ * H_ * S_ * HD_);
  unsigned short* kbuf   = take((size_t)B_ * H_ * S_ * HD_);
  unsigned short* vbuf   = take((size_t)B_ * H_ * S_ * HD_);
  unsigned short* ybuf   = take((size_t)B_ * S_ * D_);

  int nx = B_ * S_ * D_;
  cvt_bf16<<<(nx + 255) / 256, 256, 0, stream>>>(x, xb, nx);
  int nw = D_ * 3 * D_;
  cvt_bf16<<<(nw + 255) / 256, 256, 0, stream>>>(Wqkv, wqkvb, nw);
  int npj = D_ * D_;
  cvt_bf16<<<(npj + 255) / 256, 256, 0, stream>>>(Wproj, wprojb, npj);

  // QKV: [8192,1024] @ [1024,3072]
  gemm_bf16<<<dim3(32, 24), 256, 0, stream>>>(xb, wqkvb, bqkv, 3072, 0,
                                              qbuf, kbuf, vbuf, nullptr, nullptr);
  // attention: 32 q-tiles x (B*H = 64)
  attn<<<dim3(32, 64), 128, 0, stream>>>(qbuf, kbuf, vbuf, pad, ybuf);
  // proj: [8192,1024] @ [1024,1024] -> f32 out
  gemm_bf16<<<dim3(32, 8), 256, 0, stream>>>(ybuf, wprojb, bproj, 1024, 1,
                                             nullptr, nullptr, nullptr, out, pad);
}
